// RotaryAttention_16080357556670
// MI455X (gfx1250) — matrix operand order repaired and checked by reading, not compiled, not run on hardware
//
#include <hip/hip_runtime.h>
#include <hip/hip_fp16.h>

// RotaryAttention for MI455X (gfx1250, wave32, WMMA 16x16x32 f16).
//   k0a: f32 -> f16 conversion of x
//   k0b: tiled transpose + f32->f16 of Wq,Wk,Wv,Wo (B-operands K-innermost)
//   k1:  QKV projection, 64x64 macro-tile/wave (16 WMMA per K-step), async
//        double-buffered LDS staging; bias + RoPE + q*HD^-0.5 epilogue.
//        Q,K stored [B][H][S][HD]; V stored transposed [B][H][HD][S]
//   k2:  flash attention per (b,h,16-q tile), async double-buffered K/V tiles
//   k3:  output projection, 64x64 macro-tile/wave -> f32 d_out

typedef __attribute__((ext_vector_type(16))) _Float16 v16h;
typedef __attribute__((ext_vector_type(8)))  float    v8f;
typedef int v4i __attribute__((vector_size(16)));  // matches async-LDS builtin

#define B_  8
#define S_  512
#define D_  1024
#define H_  16
#define HD_ 64

#if defined(__gfx1250__) &&                                            \
    __has_builtin(__builtin_amdgcn_global_load_async_to_lds_b128) &&   \
    __has_builtin(__builtin_amdgcn_s_wait_asynccnt)
#define ASYNC_OK 1
#else
#define ASYNC_OK 0
#endif

union FragH {          // one WMMA f16 operand fragment = 32B = two float4
  v16h   h;
  float4 f[2];
};
union Pack8H {         // 8 packed f16 = 16B
  _Float16 hv[8];
  float4   f;
};

static __device__ __forceinline__ v8f wmma_f16(v16h a, v16h b, v8f c) {
  // (neg_a, A, neg_b, B, c_mod, C, reuse_a, reuse_b)
  return __builtin_amdgcn_wmma_f32_16x16x32_f16(false, a, false, b, (short)0, c,
                                                false, false);
}

// Copy 16B global -> LDS: async (CDNA5 GLOBAL_LOAD_ASYNC_TO_LDS_B128) if the
// toolchain declares the builtin, else synchronous b128 load/store.
// Builtin signature (from clang diagnostic): (v4i AS1* gsrc, v4i AS3* ldst, Ii, Ii)
static __device__ __forceinline__ void stage16(const _Float16* g, _Float16* l) {
#if ASYNC_OK
  __builtin_amdgcn_global_load_async_to_lds_b128(
      (__attribute__((address_space(1))) v4i*)g,
      (__attribute__((address_space(3))) v4i*)l, 0, 0);
#else
  *(float4*)l = *(const float4*)g;
#endif
}
static __device__ __forceinline__ void wait_staging() {
#if ASYNC_OK
  __builtin_amdgcn_s_wait_asynccnt(0);
#endif
}

// A fragment: 16x32 f16 rows; lane (n, half): elems 0..7 = k half*8..,
// elems 8..15 = k 16+half*8..  -> two contiguous b128 loads.
static __device__ __forceinline__ v16h load_a_frag(const _Float16* row, int half) {
  FragH a;
  a.f[0] = *(const float4*)(row + half * 8);
  a.f[1] = *(const float4*)(row + 16 + half * 8);
  return a.h;
}

// B fragment: 32x16 f16 stored [n][k] (K innermost): elems 0..7 = k half*8..,
// elems 8..15 = k 16+half*8.. (same k order as A).
static __device__ __forceinline__ v16h load_b_frag_kin(const _Float16* nrow, int half) {
  FragH b;
  b.f[0] = *(const float4*)(nrow + half * 8);
  b.f[1] = *(const float4*)(nrow + 16 + half * 8);
  return b.h;
}

// ---------------------------------------------------------------------------
// Kernel 0a: elementwise f32 -> f16 (x)
// ---------------------------------------------------------------------------
__global__ void cvt_f32_to_f16(const float* __restrict__ src,
                               _Float16* __restrict__ dst, int n) {
  int i = blockIdx.x * blockDim.x + threadIdx.x;
  int stride = gridDim.x * blockDim.x;
  for (; i < n; i += stride) dst[i] = (_Float16)src[i];
}

// ---------------------------------------------------------------------------
// Kernel 0b: tiled transpose + convert for the four DxD weights.
// grid = (D/32, D/32, 4), block = 256.  T[c][r] = (f16)W[r][c]
// ---------------------------------------------------------------------------
__global__ __launch_bounds__(256)
void transpose_cvt_kernel(const float* __restrict__ W0, const float* __restrict__ W1,
                          const float* __restrict__ W2, const float* __restrict__ W3,
                          _Float16* __restrict__ T0, _Float16* __restrict__ T1,
                          _Float16* __restrict__ T2, _Float16* __restrict__ T3) {
  const float* W = (blockIdx.z == 0) ? W0 : (blockIdx.z == 1) ? W1
                 : (blockIdx.z == 2) ? W2 : W3;
  _Float16* T = (blockIdx.z == 0) ? T0 : (blockIdx.z == 1) ? T1
              : (blockIdx.z == 2) ? T2 : T3;

  __shared__ _Float16 tile[32][33];
  const int c0 = blockIdx.x * 32, r0 = blockIdx.y * 32;
  const int tx = threadIdx.x & 31, ty = threadIdx.x >> 5;  // ty: 0..7
#pragma unroll
  for (int i = 0; i < 4; ++i) {
    int r = ty + i * 8;
    tile[r][tx] = (_Float16)W[(size_t)(r0 + r) * D_ + c0 + tx];
  }
  __syncthreads();
#pragma unroll
  for (int i = 0; i < 4; ++i) {
    int r = ty + i * 8;
    T[(size_t)(c0 + r) * D_ + r0 + tx] = tile[tx][r];
  }
}

// ---------------------------------------------------------------------------
// Kernel 1: QKV projection + RoPE.  grid = (M/64, H, 3), block = 32 (1 wave).
// 64 rows x 64 cols per wave: 4 A-frags x 4 B-frags -> 16 WMMA per K-step.
// W^T tile double-buffered in LDS via async loads.
// ---------------------------------------------------------------------------
__global__ __launch_bounds__(32)
void qkv_rope_kernel(const _Float16* __restrict__ xh,
                     const _Float16* __restrict__ WqT, const float* __restrict__ bq,
                     const _Float16* __restrict__ WkT, const float* __restrict__ bk,
                     const _Float16* __restrict__ WvT, const float* __restrict__ bv,
                     _Float16* __restrict__ Qd, _Float16* __restrict__ Kd,
                     _Float16* __restrict__ Vt) {
  const int lane  = threadIdx.x;
  const int mt    = blockIdx.x;   // 64-row tile of (B*S)
  const int h     = blockIdx.y;   // head -> output cols [h*64, h*64+64)
  const int which = blockIdx.z;   // 0=q 1=k 2=v

  const _Float16* WT  = (which == 0) ? WqT : (which == 1) ? WkT : WvT;
  const float*   bias = (which == 0) ? bq  : (which == 1) ? bk  : bv;

  __shared__ _Float16 Wt[2][64 * 32];  // double-buffered W^T tile [n(64)][k(32)]

  const int n    = lane & 15;
  const int half = lane >> 4;
  const _Float16* WTbase = WT + (size_t)(h * 64) * D_;

  v8f acc[4][4] = {};  // [row subtile][col subtile]

  auto stageW = [&](int buf, int k0) {
    const _Float16* g0 = WTbase + (size_t)lane * D_ + k0;
    const _Float16* g1 = WTbase + (size_t)(lane + 32) * D_ + k0;
    _Float16* l0 = &Wt[buf][lane * 32];
    _Float16* l1 = &Wt[buf][(lane + 32) * 32];
#pragma unroll
    for (int i = 0; i < 4; ++i) {
      stage16(g0 + i * 8, l0 + i * 8);
      stage16(g1 + i * 8, l1 + i * 8);
    }
  };

  int buf = 0;
  stageW(0, 0);
  wait_staging();

  for (int k0 = 0; k0 < D_; k0 += 32) {
    if (k0 + 32 < D_) stageW(buf ^ 1, k0 + 32);  // prefetch next tile into LDS

    v16h a[4];
#pragma unroll
    for (int rt = 0; rt < 4; ++rt)
      a[rt] = load_a_frag(xh + (size_t)(mt * 64 + rt * 16 + n) * D_ + k0, half);

#pragma unroll
    for (int t = 0; t < 4; ++t) {
      v16h bf = load_b_frag_kin(&Wt[buf][(t * 16 + n) * 32], half);
#pragma unroll
      for (int rt = 0; rt < 4; ++rt)
        acc[rt][t] = wmma_f16(a[rt], bf, acc[rt][t]);
    }
    wait_staging();
    buf ^= 1;
  }

  // Epilogue
  if (which == 2) {
    // V: bias, store transposed [B][H][HD][S] as packed b128 along S
#pragma unroll
    for (int rt = 0; rt < 4; ++rt) {
      int row0 = mt * 64 + rt * 16 + half * 8;
      int b = row0 >> 9, s0v = row0 & (S_ - 1);
#pragma unroll
      for (int t = 0; t < 4; ++t) {
        int c = t * 16 + n;
        float bb = bias[h * HD_ + c];
        Pack8H pk;
#pragma unroll
        for (int j = 0; j < 8; ++j) pk.hv[j] = (_Float16)(acc[rt][t][j] + bb);
        *(float4*)(Vt + (((size_t)b * H_ + h) * HD_ + c) * S_ + s0v) = pk.f;
      }
    }
  } else {
    // Q/K: bias + RoPE (+ q scale), store [B][H][S][HD]
    float invfreq[2];
#pragma unroll
    for (int t = 0; t < 2; ++t) {
      // 10000^(-f/32) = exp(-f * ln(10000)/32)
      invfreq[t] = __expf((float)(n + t * 16) * -0.28782313662425572f);
    }
    const float qscale = 0.125f;  // HD^-0.5
    _Float16* Out = (which == 0) ? Qd : Kd;
#pragma unroll
    for (int rt = 0; rt < 4; ++rt) {
      int row0 = mt * 64 + rt * 16 + half * 8;
      int b = row0 >> 9, s0v = row0 & (S_ - 1);
#pragma unroll
      for (int j = 0; j < 8; ++j) {
        int s = s0v + j;
        float pos = (float)s;
#pragma unroll
        for (int t = 0; t < 2; ++t) {
          int c = t * 16 + n;  // < 32; pairs with c+32
          float x1 = acc[rt][t][j]     + bias[h * HD_ + c];
          float x2 = acc[rt][t + 2][j] + bias[h * HD_ + c + 32];
          float ang = pos * invfreq[t];
          float sa = __sinf(ang), ca = __cosf(ang);
          float outl = x1 * ca - x2 * sa;
          float outh = x2 * ca + x1 * sa;
          if (which == 0) { outl *= qscale; outh *= qscale; }
          size_t base = (((size_t)b * H_ + h) * S_ + s) * HD_;
          Out[base + c]      = (_Float16)outl;
          Out[base + c + 32] = (_Float16)outh;
        }
      }
    }
  }
}

// ---------------------------------------------------------------------------
// Kernel 2: flash attention.  grid = (S/16, H, B), block = 32 (1 wave).
// Q,K: [B][H][S][HD]; V: [B][H][HD][S].  K/V tiles double-buffered via async.
// ---------------------------------------------------------------------------
__global__ __launch_bounds__(32)
void flash_attn_kernel(const _Float16* __restrict__ Q,
                       const _Float16* __restrict__ K,
                       const _Float16* __restrict__ Vt,
                       _Float16* __restrict__ A /* [B*S][D] */) {
  const int lane = threadIdx.x;
  const int qt   = blockIdx.x;
  const int h    = blockIdx.y;
  const int b    = blockIdx.z;
  const int n    = lane & 15;
  const int half = lane >> 4;

  const size_t hoff = ((size_t)b * H_ + h) * (size_t)S_ * HD_;
  const _Float16* Qh = Q  + hoff;
  const _Float16* Kh = K  + hoff;
  const _Float16* Vh = Vt + hoff;  // [HD][S]

  __shared__ _Float16 Qs[16 * 64];       // [q][hd]
  __shared__ _Float16 Ks[2][32 * 64];    // [key][hd]  (B of QK^T, k-innermost)
  __shared__ _Float16 VsT[2][64 * 32];   // [hd][key]  (B of PV,   k-innermost)
  __shared__ _Float16 Ps[16 * 32];       // [q][key]   (A of PV)

  auto stageKV = [&](int buf, int kt) {
    const _Float16* kg = Kh + (size_t)(kt * 32 + lane) * HD_;
    _Float16* kl = &Ks[buf][lane * 64];
#pragma unroll
    for (int i = 0; i < 8; ++i) stage16(kg + i * 8, kl + i * 8);

    const _Float16* v0 = Vh + (size_t)lane * S_ + kt * 32;
    const _Float16* v1 = Vh + (size_t)(lane + 32) * S_ + kt * 32;
    _Float16* l0 = &VsT[buf][lane * 32];
    _Float16* l1 = &VsT[buf][(lane + 32) * 32];
#pragma unroll
    for (int i = 0; i < 4; ++i) {
      stage16(v0 + i * 8, l0 + i * 8);
      stage16(v1 + i * 8, l1 + i * 8);
    }
  };

  // Stage Q tile (each lane: half a row = 4 x 16B) + first K/V tile
  {
    const _Float16* qg = Qh + (size_t)(qt * 16 + (lane >> 1)) * HD_ + (lane & 1) * 32;
    _Float16* ql = Qs + (lane >> 1) * 64 + (lane & 1) * 32;
#pragma unroll
    for (int i = 0; i < 4; ++i) stage16(qg + i * 8, ql + i * 8);
  }
  stageKV(0, 0);
  wait_staging();

  const v16h aQ0 = load_a_frag(Qs + n * 64, half);       // hd 0..31
  const v16h aQ1 = load_a_frag(Qs + n * 64 + 32, half);  // hd 32..63

  v8f o[4] = {};
  float mi[8], li[8];
#pragma unroll
  for (int j = 0; j < 8; ++j) { mi[j] = -1e30f; li[j] = 0.0f; }

  int buf = 0;
  for (int kt = 0; kt < S_ / 32; ++kt) {
    if (kt + 1 < S_ / 32) stageKV(buf ^ 1, kt + 1);

    // Logits: two 16x16 tiles (keys u*16..); B = K^T contiguous in Ks rows
    v8f sacc[2];
#pragma unroll
    for (int u = 0; u < 2; ++u) {
      v16h bk0 = load_b_frag_kin(&Ks[buf][(u * 16 + n) * 64], half);       // hd 0..31
      v16h bk1 = load_b_frag_kin(&Ks[buf][(u * 16 + n) * 64 + 32], half);  // hd 32..63
      v8f z = {};
      z = wmma_f16(aQ0, bk0, z);
      sacc[u] = wmma_f16(aQ1, bk1, z);
    }

    // Online softmax; per-row reduction across the 16-lane group
#pragma unroll
    for (int j = 0; j < 8; ++j) {
      float s0 = sacc[0][j] * 0.125f;  // second HD^-0.5 (reference scales twice)
      float s1 = sacc[1][j] * 0.125f;
      float tmax = fmaxf(s0, s1);
#pragma unroll
      for (int d = 1; d < 16; d <<= 1) tmax = fmaxf(tmax, __shfl_xor(tmax, d, 16));
      float mnew = fmaxf(mi[j], tmax);
      float p0 = __expf(s0 - mnew);
      float p1 = __expf(s1 - mnew);
      float rs = p0 + p1;
#pragma unroll
      for (int d = 1; d < 16; d <<= 1) rs += __shfl_xor(rs, d, 16);
      float alpha = __expf(mi[j] - mnew);
      li[j] = li[j] * alpha + rs;
      mi[j] = mnew;
#pragma unroll
      for (int t = 0; t < 4; ++t) o[t][j] *= alpha;
      int m = j + half * 8;  // C-layout row
      Ps[m * 32 + n]      = (_Float16)p0;
      Ps[m * 32 + 16 + n] = (_Float16)p1;
    }

    // P(16x32) x V(32x64); LDS ops of one wave are in-order, so the Ps
    // stores above are visible to these loads.
    v16h aP = load_a_frag(Ps + n * 32, half);
#pragma unroll
    for (int t = 0; t < 4; ++t) {
      v16h bv_ = load_b_frag_kin(&VsT[buf][(t * 16 + n) * 32], half);
      o[t] = wmma_f16(aP, bv_, o[t]);
    }
    wait_staging();
    buf ^= 1;
  }

  // Normalize and store row-major [B*S][D] (cols h*64+hd)
#pragma unroll
  for (int j = 0; j < 8; ++j) {
    int m = j + half * 8;
    size_t row = (size_t)b * S_ + qt * 16 + m;
    float inv = 1.0f / li[j];
#pragma unroll
    for (int t = 0; t < 4; ++t) {
      A[row * D_ + h * HD_ + t * 16 + n] = (_Float16)(o[t][j] * inv);
    }
  }
}

// ---------------------------------------------------------------------------
// Kernel 3: output projection.  grid = (M/64, D/64), block = 32 (1 wave).
// 64x64 macro-tile, double-buffered async W^T staging.
// ---------------------------------------------------------------------------
__global__ __launch_bounds__(32)
void out_proj_kernel(const _Float16* __restrict__ Ain,
                     const _Float16* __restrict__ WoT,
                     const float* __restrict__ bo, float* __restrict__ out) {
  const int lane = threadIdx.x;
  const int mt   = blockIdx.x;
  const int nt   = blockIdx.y;  // 64-col chunk
  const int n    = lane & 15;
  const int half = lane >> 4;

  __shared__ _Float16 Wt[2][64 * 32];
  const _Float16* WTbase = WoT + (size_t)(nt * 64) * D_;

  v8f acc[4][4] = {};

  auto stageW = [&](int buf, int k0) {
    const _Float16* g0 = WTbase + (size_t)lane * D_ + k0;
    const _Float16* g1 = WTbase + (size_t)(lane + 32) * D_ + k0;
    _Float16* l0 = &Wt[buf][lane * 32];
    _Float16* l1 = &Wt[buf][(lane + 32) * 32];
#pragma unroll
    for (int i = 0; i < 4; ++i) {
      stage16(g0 + i * 8, l0 + i * 8);
      stage16(g1 + i * 8, l1 + i * 8);
    }
  };

  int buf = 0;
  stageW(0, 0);
  wait_staging();

  for (int k0 = 0; k0 < D_; k0 += 32) {
    if (k0 + 32 < D_) stageW(buf ^ 1, k0 + 32);

    v16h a[4];
#pragma unroll
    for (int rt = 0; rt < 4; ++rt)
      a[rt] = load_a_frag(Ain + (size_t)(mt * 64 + rt * 16 + n) * D_ + k0, half);

#pragma unroll
    for (int t = 0; t < 4; ++t) {
      v16h bf = load_b_frag_kin(&Wt[buf][(t * 16 + n) * 32], half);
#pragma unroll
      for (int rt = 0; rt < 4; ++rt)
        acc[rt][t] = wmma_f16(a[rt], bf, acc[rt][t]);
    }
    wait_staging();
    buf ^= 1;
  }

#pragma unroll
  for (int rt = 0; rt < 4; ++rt) {
#pragma unroll
    for (int j = 0; j < 8; ++j) {
      size_t row = (size_t)mt * 64 + rt * 16 + j + half * 8;
#pragma unroll
      for (int t = 0; t < 4; ++t) {
        int col = nt * 64 + t * 16 + n;
        out[row * D_ + col] = acc[rt][t][j] + bo[col];
      }
    }
  }
}

// ---------------------------------------------------------------------------
extern "C" void kernel_launch(void* const* d_in, const int* in_sizes, int n_in,
                              void* d_out, int out_size, void* d_ws, size_t ws_size,
                              hipStream_t stream) {
  const float* x  = (const float*)d_in[0];
  const float* Wq = (const float*)d_in[1];
  const float* bq = (const float*)d_in[2];
  const float* Wk = (const float*)d_in[3];
  const float* bk = (const float*)d_in[4];
  const float* Wv = (const float*)d_in[5];
  const float* bv = (const float*)d_in[6];
  const float* Wo = (const float*)d_in[7];
  const float* bo = (const float*)d_in[8];
  float* out = (float*)d_out;

  const size_t NX = (size_t)B_ * S_ * D_;  // 4M elements
  const size_t NW = (size_t)D_ * D_;       // 1M elements

  char* ws = (char*)d_ws;
  _Float16* xh  = (_Float16*)ws; ws += NX * 2;
  _Float16* WqT = (_Float16*)ws; ws += NW * 2;
  _Float16* WkT = (_Float16*)ws; ws += NW * 2;
  _Float16* WvT = (_Float16*)ws; ws += NW * 2;
  _Float16* WoT = (_Float16*)ws; ws += NW * 2;
  _Float16* Qd  = (_Float16*)ws; ws += NX * 2;
  _Float16* Kd  = (_Float16*)ws; ws += NX * 2;
  _Float16* Vt  = (_Float16*)ws; ws += NX * 2;  // [B][H][HD][S]
  _Float16* Ab  = (_Float16*)ws; ws += NX * 2;

  cvt_f32_to_f16<<<2048, 256, 0, stream>>>(x, xh, (int)NX);
  transpose_cvt_kernel<<<dim3(D_ / 32, D_ / 32, 4), 256, 0, stream>>>(
      Wq, Wk, Wv, Wo, WqT, WkT, WvT, WoT);

  qkv_rope_kernel<<<dim3((B_ * S_) / 64, H_, 3), 32, 0, stream>>>(
      xh, WqT, bq, WkT, bk, WvT, bv, Qd, Kd, Vt);

  flash_attn_kernel<<<dim3(S_ / 16, H_, B_), 32, 0, stream>>>(Qd, Kd, Vt, Ab);

  out_proj_kernel<<<dim3((B_ * S_) / 64, D_ / 64), 32, 0, stream>>>(Ab, WoT, bo, out);
}
